// DenseGATv2_7069516169275
// MI455X (gfx1250) — matrix operand order repaired and checked by reading, not compiled, not run on hardware
//
#include <hip/hip_runtime.h>
#include <hip/hip_bf16.h>

typedef __attribute__((ext_vector_type(16))) _Float16 v16h;
typedef __attribute__((ext_vector_type(8)))  _Float16 v8h;
typedef __attribute__((ext_vector_type(8)))  float    v8f;

#define NB 1024
#define ND 128
#define NH 8
#define NQ 16
#define NEG_SLOPE 0.15f
#define MASK_VAL -1000000000.0f
#define LOG2E 1.4426950408889634f

#if __has_builtin(__builtin_amdgcn_exp2f)
#define EXP2F(x) __builtin_amdgcn_exp2f(x)
#else
#define EXP2F(x) exp2f(x)
#endif

// ---------------------------------------------------------------------------
// Kernel 1: per (b,n) row — stage h row in LDS, compute:
//   vT[b,h,q,n] = f16( h[b,n,:] @ Wv[:, h*16+q] )   (transposed: WMMA B-operand)
//   s_i[b,h,n]  = h[b,n,:] @ Wa[0:128,  h]
//   s_j[b,h,n]  = h[b,n,:] @ Wa[128:256,h]
// grid: B*N blocks of 128 threads
// ---------------------------------------------------------------------------
__global__ void gat_proj_kernel(const float* __restrict__ hmat,
                                const float* __restrict__ Wa,
                                const float* __restrict__ Wv,
                                float* __restrict__ s_i,
                                float* __restrict__ s_j,
                                _Float16* __restrict__ vT) {
    __shared__ float hrow[ND];
    const int bn = blockIdx.x;           // b*N + n
    const int b  = bn >> 10;
    const int n  = bn & (NB - 1);
    const int t  = threadIdx.x;          // 0..127

    hrow[t] = hmat[bn * ND + t];
    __syncthreads();

    float acc = 0.0f;
#pragma unroll 8
    for (int d = 0; d < ND; ++d)
        acc = fmaf(hrow[d], Wv[d * (NH * NQ) + t], acc);
    const int hh = t >> 4, q = t & 15;
    vT[((b * NH + hh) * NQ + q) * NB + n] = (_Float16)acc;

    if (t < 16) {
        const int which = t >> 3;        // 0 -> s_i, 1 -> s_j
        const int hc    = t & 7;
        const int off   = which * ND;
        float a2 = 0.0f;
#pragma unroll 8
        for (int d = 0; d < ND; ++d)
            a2 = fmaf(hrow[d], Wa[(off + d) * NH + hc], a2);
        float* dst = which ? s_j : s_i;
        dst[(b * NH + hc) * NB + n] = a2;
    }
}

// ---------------------------------------------------------------------------
// Kernel 2: fused flash-style attention per (b, h, 16-row i-tile), one wave32.
//   pass 1: compute scores once (scaled by log2e), f16-round into LDS,
//           track per-row max (over the f16-rounded values). No exp.
//   pass 2: alpha = exp2(sc - M) from LDS, accumulate softmax denominator L
//           and P@V via v_wmma_f32_16x16x32_f16 (normalize by 1/L at the end).
// grid: (N/16, H, B), block: 32
// ---------------------------------------------------------------------------
__global__ void gat_attn_kernel(const float* __restrict__ s_i,
                                const float* __restrict__ s_j,
                                const _Float16* __restrict__ vT,
                                const int* __restrict__ edge,
                                const float* __restrict__ Wa,
                                const int* __restrict__ local_only,
                                float* __restrict__ attn) {
    // +8 f16 row padding: row stride 516 dwords -> LDS banks rotate by 4/row
    __shared__ _Float16 sc_sm[16][NB + 8];
    __shared__ float    rl_sm[16];

    const int lane = threadIdx.x;        // 0..31
    const int b  = blockIdx.z;
    const int hh = blockIdx.y;
    const int i0 = blockIdx.x * 16;

    const int row = lane & 15;           // A row, and B/D's N index (q)
    const int khalf = lane >> 4;         // which K-half this lane covers
    const int klo = khalf ? 8 : 0;
    const int irow = i0 + row;

    const float si = s_i[(b * NH + hh) * NB + irow] ;
    const float we = Wa[2 * ND * NH + hh];
    const int local = local_only[0];

    const float* sjp = s_j + (b * NH + hh) * NB;
    const int*   ep  = edge + (b * NB + irow) * NB;

    // ---- pass 1: scores -> LDS (f16, log2e-scaled), per-row running max
    float m = -3.0e38f;
    for (int j0 = 0; j0 < NB; j0 += 32) {
        __builtin_prefetch(ep + j0 + 256, 0, 1);
        const int jA = j0 + klo;
#pragma unroll
        for (int g = 0; g < 2; ++g) {
            const int jb = jA + g * 16;
            const float4* s4 = (const float4*)(sjp + jb);
            const int4*   e4 = (const int4*)(ep + jb);
            const float4 sa = s4[0], sb = s4[1];
            const int4   ea = e4[0], eb = e4[1];
            const float vs[8] = {sa.x, sa.y, sa.z, sa.w, sb.x, sb.y, sb.z, sb.w};
            const int   ve[8] = {ea.x, ea.y, ea.z, ea.w, eb.x, eb.y, eb.z, eb.w};
            v8h hv;
#pragma unroll
            for (int e = 0; e < 8; ++e) {
                float x = si + vs[e] + (ve[e] ? we : 0.0f);
                x = (x >= 0.0f) ? x : NEG_SLOPE * x;       // leaky_relu
                if (local && !ve[e]) x = MASK_VAL;          // local_only mask
                x *= LOG2E;                                 // exp -> exp2
                x = fminf(fmaxf(x, -60000.0f), 60000.0f);   // keep f16-finite
                const _Float16 hx = (_Float16)x;
                hv[e] = hx;
                m = fmaxf(m, (float)hx);                    // max of rounded vals
            }
            *(v8h*)&sc_sm[row][jb] = hv;
        }
    }
    const float M = fmaxf(m, __shfl_xor(m, 16, 32));        // combine K-halves
    __syncthreads();

    // ---- pass 2: alpha from LDS, denominator + WMMA accumulate
    v8f c = {0.f, 0.f, 0.f, 0.f, 0.f, 0.f, 0.f, 0.f};
    float lsum = 0.0f;
    const _Float16* vrow = vT + ((b * NH + hh) * NQ + row) * NB + khalf * 16;
#pragma unroll 2
    for (int j0 = 0; j0 < NB; j0 += 32) {
        __builtin_prefetch(vrow + j0 + 256, 0, 1);
        const v8h slo = *(const v8h*)&sc_sm[row][j0 + khalf * 16];
        const v8h shi = *(const v8h*)&sc_sm[row][j0 + khalf * 16 + 8];
        v16h a;
#pragma unroll
        for (int e = 0; e < 8; ++e) {
            const float f0 = EXP2F((float)slo[e] - M);      // <= 1
            const float f1 = EXP2F((float)shi[e] - M);
            a[e]     = (_Float16)f0;
            a[e + 8] = (_Float16)f1;
            lsum += f0 + f1;
        }
        const v16h bv = *(const v16h*)(vrow + j0);
        c = __builtin_amdgcn_wmma_f32_16x16x32_f16(
                /*neg_a=*/false, a, /*neg_b=*/false, bv,
                /*c_mod=*/(short)0, c, /*reuse_a=*/false, /*reuse_b=*/false);
    }
    lsum += __shfl_xor(lsum, 16, 32);                       // row denominator L
    if (lane < 16) rl_sm[lane] = 1.0f / lsum;
    __syncthreads();

    // ---- normalize rows by 1/L and store to attn (B, N, H*16)
    const int mb = khalf * 8;                               // C/D rows this lane
    float* op = attn + ((b * NB + i0) * NH + hh) * NQ + row;
#pragma unroll
    for (int e = 0; e < 8; ++e)
        op[(mb + e) * (NH * NQ)] = c[e] * rl_sm[mb + e];
}

// ---------------------------------------------------------------------------
// Kernel 3: out = attn(B*N,128) @ Wo(128,128), LDS-staged fp32
// grid: B*N blocks of 128 threads
// ---------------------------------------------------------------------------
__global__ void gat_outproj_kernel(const float* __restrict__ attn,
                                   const float* __restrict__ Wo,
                                   float* __restrict__ out) {
    __shared__ float arow[ND];
    const int bn = blockIdx.x;
    const int t  = threadIdx.x;
    arow[t] = attn[bn * ND + t];
    __syncthreads();
    float acc = 0.0f;
#pragma unroll 8
    for (int d = 0; d < ND; ++d)
        acc = fmaf(arow[d], Wo[d * ND + t], acc);
    out[bn * ND + t] = acc;
}

// ---------------------------------------------------------------------------
extern "C" void kernel_launch(void* const* d_in, const int* in_sizes, int n_in,
                              void* d_out, int out_size, void* d_ws, size_t ws_size,
                              hipStream_t stream) {
    const float* h          = (const float*)d_in[0];
    const int*   same       = (const int*)d_in[1];
    const float* Wa         = (const float*)d_in[2];
    const float* Wv         = (const float*)d_in[3];
    const float* Wo         = (const float*)d_in[4];
    const int*   local_only = (const int*)d_in[5];
    float*       out        = (float*)d_out;

    char* ws = (char*)d_ws;
    float*     s_i  = (float*)(ws);                       // B*H*N f32    = 128 KB
    float*     s_j  = (float*)(ws + 131072);              // B*H*N f32    = 128 KB
    _Float16*  vT   = (_Float16*)(ws + 262144);           // B*H*16*N f16 = 1 MB
    float*     attn = (float*)(ws + 1310720);             // B*N*128 f32  = 2 MB

    gat_proj_kernel<<<4 * NB, 128, 0, stream>>>(h, Wa, Wv, s_i, s_j, vT);

    gat_attn_kernel<<<dim3(NB / 16, NH, 4), 32, 0, stream>>>(
        s_i, s_j, vT, same, Wa, local_only, attn);

    gat_outproj_kernel<<<4 * NB, 128, 0, stream>>>(attn, Wo, out);
}
